// Backbone_47390669144486
// MI455X (gfx1250) — matrix operand order repaired and checked by reading, not compiled, not run on hardware
//
#include <hip/hip_runtime.h>
#include <hip/hip_bf16.h>
#include <math.h>

// ---------------------------------------------------------------------------
// Problem constants (from reference): way=5, nq=75, C=64, H=W=20 -> P=400
// ---------------------------------------------------------------------------
#define WAY   5
#define NQ    75
#define CCH   64
#define PP    400
#define TILES 25          // PP / 16
#define TEMP_ATTN 5.0f

typedef __attribute__((ext_vector_type(16))) _Float16 v16h;
typedef __attribute__((ext_vector_type(8)))  _Float16 v8h;
typedef __attribute__((ext_vector_type(8)))  float    v8f;

// ---------------------------------------------------------------------------
// Kernel 1: per-image preprocessing.
//   x_n = x - mean_c(x);  y = relu(bn(conv1x1(x_n)));  y = y / max(||y||_c,1e-8)
// Output layout: featT[n][p][c]  (f16, c contiguous) -- WMMA-friendly.
//   n in [0,80): n<5 -> support image n, else query image n-5.
// ---------------------------------------------------------------------------
__global__ __launch_bounds__(256) void preproc_kernel(
    const float* __restrict__ spt, const float* __restrict__ qry,
    const float* __restrict__ conv_w, const float* __restrict__ bn_g,
    const float* __restrict__ bn_b,  const float* __restrict__ bn_m,
    const float* __restrict__ bn_v,  _Float16* __restrict__ featT)
{
    __shared__ float Wl[CCH * CCH];
    __shared__ float sc[CCH], bb[CCH], mm[CCH];
    const int tid = threadIdx.x;
    for (int i = tid; i < CCH * CCH; i += 256) Wl[i] = conv_w[i];
    if (tid < CCH) {
        sc[tid] = bn_g[tid] * rsqrtf(bn_v[tid] + 1e-5f);
        bb[tid] = bn_b[tid];
        mm[tid] = bn_m[tid];
    }
    __syncthreads();

    const int n = blockIdx.x;
    const float* X = (n < WAY) ? (spt + (size_t)n * CCH * PP)
                               : (qry + (size_t)(n - WAY) * CCH * PP);

    for (int p = tid; p < PP; p += 256) {
        float xc[CCH];
        float mu = 0.0f;
        for (int c = 0; c < CCH; ++c) { xc[c] = X[c * PP + p]; mu += xc[c]; }
        mu *= (1.0f / (float)CCH);
        for (int c = 0; c < CCH; ++c) xc[c] -= mu;

        float yb[CCH];
        float nrm = 0.0f;
        for (int o = 0; o < CCH; ++o) {
            float acc = 0.0f;
            const float* wr = &Wl[o * CCH];
            for (int c = 0; c < CCH; ++c) acc += wr[c] * xc[c];
            float y = (acc - mm[o]) * sc[o] + bb[o];
            y = fmaxf(y, 0.0f);
            yb[o] = y;
            nrm += y * y;
        }
        float inv = 1.0f / fmaxf(sqrtf(nrm), 1e-8f);
        _Float16* dst = featT + ((size_t)n * PP + p) * CCH;
        for (int o = 0; o < CCH; ++o) dst[o] = (_Float16)(yb[o] * inv);
    }
}

// ---------------------------------------------------------------------------
// WMMA fragment loaders (contiguous k order, same in A and B, wave32).
//  A (16x32, MxK): lane<16 holds M=lane, K = {k0..k0+15}, k0=0
//                  lane>=16 holds M=lane-16, same with k0=16.
//  B (32x16, KxN): lane holds N=lane%16, K = {cb..cb+15}, cb = lane<16 ? 0 : 16.
//  featT rows are [pos][c] with c contiguous -> aligned 16B/32B chunks.
// ---------------------------------------------------------------------------
__device__ __forceinline__ v16h load_a(const _Float16* __restrict__ S,
                                       int mi, int kk, int lane)
{
    const int row = mi * 16 + (lane & 15);
    const int c0  = kk * 32 + ((lane < 16) ? 0 : 16);
    const _Float16* p = S + (size_t)row * CCH + c0;
    v8h lo = *(const v8h*)(p);
    v8h hi = *(const v8h*)(p + 8);
    v16h a;
#pragma unroll
    for (int j = 0; j < 8; ++j) { a[j] = lo[j]; a[j + 8] = hi[j]; }
    return a;
}

__device__ __forceinline__ v16h load_b(const _Float16* __restrict__ Q,
                                       int ni, int kk, int lane)
{
    const int col = ni * 16 + (lane & 15);
    const int c0  = kk * 32 + ((lane < 16) ? 0 : 16);
    return *(const v16h*)(Q + (size_t)col * CCH + c0);
}

// X_tile(16x16) = A(16x64) * B(64x16) via two 16x16x32 f16 WMMAs.
__device__ __forceinline__ v8f tile_mm(v16h a0, v16h a1,
                                       const _Float16* __restrict__ Q,
                                       int ni, int lane)
{
    v8f acc = {};
    v16h b0 = load_b(Q, ni, 0, lane);
    v16h b1 = load_b(Q, ni, 1, lane);
    acc = __builtin_amdgcn_wmma_f32_16x16x32_f16(false, a0, false, b0,
                                                 (short)0, acc, false, false);
    acc = __builtin_amdgcn_wmma_f32_16x16x32_f16(false, a1, false, b1,
                                                 (short)0, acc, false, false);
    return acc;
}

// ---------------------------------------------------------------------------
// Kernel 2: one block per (q, w) pair.  X[i,k] = <s_w[:,i], q_q[:,k]>, 400x400.
//  Pass 1: col/row sums + sums of squares  -> mean, 1/(std*T) per col/row.
//  Pass 2: softmax denominators (shift-invariance: subtract mean, / (std*T)).
//  Pass 3: attnS[i] = sum_k z_s/colDen[k];  attnQ[k] = sum_i z_q/rowDen[i].
//  Then attention-weighted pooling of mean-subtracted raw features + cosine.
// ---------------------------------------------------------------------------
__global__ __launch_bounds__(256) void corr_attn_kernel(
    const float* __restrict__ spt, const float* __restrict__ qry,
    const _Float16* __restrict__ featT, const float* __restrict__ scale,
    float* __restrict__ out)
{
    __shared__ float colSum[PP], colSq[PP], colMean[PP], colInv[PP], colDen[PP];
    __shared__ float rowSum[PP], rowSq[PP], rowMean[PP], rowInv[PP], rowDen[PP];
    __shared__ float attnS[PP], attnQ[PP];
    __shared__ float muS[PP], muQ[PP];
    __shared__ float sAtt[CCH], qAtt[CCH];

    const int tid  = threadIdx.x;
    const int lane = tid & 31;
    const int wave = tid >> 5;
    const int q = blockIdx.x / WAY;
    const int w = blockIdx.x % WAY;

    const _Float16* S = featT + (size_t)w * PP * CCH;
    const _Float16* Q = featT + (size_t)(WAY + q) * PP * CCH;

    for (int i = tid; i < PP; i += 256) {
        colSum[i] = 0.f; colSq[i] = 0.f; colDen[i] = 0.f;
        rowSum[i] = 0.f; rowSq[i] = 0.f; rowDen[i] = 0.f;
        attnS[i] = 0.f;  attnQ[i] = 0.f;
    }
    __syncthreads();

    // ---------------- PASS 1: statistics ----------------
    for (int mi = wave; mi < TILES; mi += 8) {
        v16h a0 = load_a(S, mi, 0, lane);
        v16h a1 = load_a(S, mi, 1, lane);
        float rs[8], rq[8];
#pragma unroll
        for (int r = 0; r < 8; ++r) { rs[r] = 0.f; rq[r] = 0.f; }
        for (int ni = 0; ni < TILES; ++ni) {
            v8f x = tile_mm(a0, a1, Q, ni, lane);
            float cs = 0.f, cq = 0.f;
#pragma unroll
            for (int r = 0; r < 8; ++r) {
                float v = x[r];
                cs += v; cq += v * v;
                rs[r] += v; rq[r] += v * v;
            }
            const int n = ni * 16 + (lane & 15);
            atomicAdd(&colSum[n], cs);
            atomicAdd(&colSq[n],  cq);
        }
#pragma unroll
        for (int r = 0; r < 8; ++r) {
#pragma unroll
            for (int off = 1; off < 16; off <<= 1) {
                rs[r] += __shfl_xor(rs[r], off, 32);
                rq[r] += __shfl_xor(rq[r], off, 32);
            }
        }
        if ((lane & 15) == 0) {
            const int base = mi * 16 + ((lane < 16) ? 0 : 8);
#pragma unroll
            for (int r = 0; r < 8; ++r) { rowSum[base + r] = rs[r]; rowSq[base + r] = rq[r]; }
        }
    }
    __syncthreads();

    // mean & 1/(std*T)   (unbiased variance, ddof=1, eps=1e-5, T=5)
    for (int i = tid; i < PP; i += 256) {
        float ms  = colSum[i] * (1.0f / PP);
        float var = (colSq[i] - (float)PP * ms * ms) * (1.0f / (PP - 1));
        colMean[i] = ms;
        colInv[i]  = 1.0f / (sqrtf(var + 1e-5f) * TEMP_ATTN);
        float mr   = rowSum[i] * (1.0f / PP);
        float varr = (rowSq[i] - (float)PP * mr * mr) * (1.0f / (PP - 1));
        rowMean[i] = mr;
        rowInv[i]  = 1.0f / (sqrtf(varr + 1e-5f) * TEMP_ATTN);
    }
    __syncthreads();

    // ---------------- PASS 2: softmax denominators ----------------
    for (int mi = wave; mi < TILES; mi += 8) {
        v16h a0 = load_a(S, mi, 0, lane);
        v16h a1 = load_a(S, mi, 1, lane);
        const int mbase = mi * 16 + ((lane < 16) ? 0 : 8);
        float rm[8], ri[8], rden[8];
#pragma unroll
        for (int r = 0; r < 8; ++r) {
            rm[r] = rowMean[mbase + r]; ri[r] = rowInv[mbase + r]; rden[r] = 0.f;
        }
        for (int ni = 0; ni < TILES; ++ni) {
            v8f x = tile_mm(a0, a1, Q, ni, lane);
            const int n = ni * 16 + (lane & 15);
            const float cm = colMean[n], ci = colInv[n];
            float cden = 0.f;
#pragma unroll
            for (int r = 0; r < 8; ++r) {
                float v = x[r];
                cden    += __expf((v - cm)    * ci);
                rden[r] += __expf((v - rm[r]) * ri[r]);
            }
            atomicAdd(&colDen[n], cden);
        }
#pragma unroll
        for (int r = 0; r < 8; ++r) {
#pragma unroll
            for (int off = 1; off < 16; off <<= 1)
                rden[r] += __shfl_xor(rden[r], off, 32);
        }
        if ((lane & 15) == 0) {
#pragma unroll
            for (int r = 0; r < 8; ++r) rowDen[mbase + r] = rden[r];
        }
    }
    __syncthreads();

    // ---------------- PASS 3: attention accumulation ----------------
    for (int mi = wave; mi < TILES; mi += 8) {
        v16h a0 = load_a(S, mi, 0, lane);
        v16h a1 = load_a(S, mi, 1, lane);
        const int mbase = mi * 16 + ((lane < 16) ? 0 : 8);
        float rm[8], ri[8], rdi[8], ras[8];
#pragma unroll
        for (int r = 0; r < 8; ++r) {
            rm[r]  = rowMean[mbase + r];
            ri[r]  = rowInv[mbase + r];
            rdi[r] = 1.0f / rowDen[mbase + r];
            ras[r] = 0.f;
        }
        for (int ni = 0; ni < TILES; ++ni) {
            v8f x = tile_mm(a0, a1, Q, ni, lane);
            const int n = ni * 16 + (lane & 15);
            const float cm = colMean[n], ci = colInv[n];
            const float cdi = 1.0f / colDen[n];
            float aq = 0.f;
#pragma unroll
            for (int r = 0; r < 8; ++r) {
                float v = x[r];
                ras[r] += __expf((v - cm)    * ci)    * cdi;
                aq     += __expf((v - rm[r]) * ri[r]) * rdi[r];
            }
            atomicAdd(&attnQ[n], aq);
        }
#pragma unroll
        for (int r = 0; r < 8; ++r) {
#pragma unroll
            for (int off = 1; off < 16; off <<= 1)
                ras[r] += __shfl_xor(ras[r], off, 32);
        }
        if ((lane & 15) == 0) {
#pragma unroll
            for (int r = 0; r < 8; ++r) attnS[mbase + r] = ras[r];
        }
    }
    __syncthreads();

    // ---------------- pooling + cosine ----------------
    const float* Sraw = spt + (size_t)w * CCH * PP;
    const float* Qraw = qry + (size_t)q * CCH * PP;
    for (int p = tid; p < PP; p += 256) {
        float s = 0.f, t = 0.f;
        for (int c = 0; c < CCH; ++c) { s += Sraw[c * PP + p]; t += Qraw[c * PP + p]; }
        muS[p] = s * (1.0f / CCH);
        muQ[p] = t * (1.0f / CCH);
    }
    __syncthreads();

    if (tid < CCH) {
        const int c = tid;
        float acc = 0.f;
        for (int p = 0; p < PP; ++p) acc += attnS[p] * (Sraw[c * PP + p] - muS[p]);
        sAtt[c] = acc * (1.0f / PP);
    } else if (tid < 2 * CCH) {
        const int c = tid - CCH;
        float acc = 0.f;
        for (int p = 0; p < PP; ++p) acc += attnQ[p] * (Qraw[c * PP + p] - muQ[p]);
        qAtt[c] = acc * (1.0f / PP);
    }
    __syncthreads();

    if (tid == 0) {
        float ss = 0.f, qq = 0.f, sq = 0.f;
        for (int c = 0; c < CCH; ++c) {
            float a = sAtt[c], b = qAtt[c];
            ss += a * a; qq += b * b; sq += a * b;
        }
        float ns  = fmaxf(sqrtf(ss), 1e-6f);
        float nq2 = fmaxf(sqrtf(qq), 1e-6f);
        out[q * WAY + w] = (sq / (ns * nq2)) * scale[0];
    }
}

// ---------------------------------------------------------------------------
// Launch
// ---------------------------------------------------------------------------
extern "C" void kernel_launch(void* const* d_in, const int* in_sizes, int n_in,
                              void* d_out, int out_size, void* d_ws, size_t ws_size,
                              hipStream_t stream)
{
    const float* spt    = (const float*)d_in[0];
    const float* qry    = (const float*)d_in[1];
    const float* conv_w = (const float*)d_in[2];
    const float* bn_g   = (const float*)d_in[3];
    const float* bn_b   = (const float*)d_in[4];
    const float* bn_m   = (const float*)d_in[5];
    const float* bn_v   = (const float*)d_in[6];
    const float* scale  = (const float*)d_in[7];

    _Float16* featT = (_Float16*)d_ws;   // (5+75) * 400 * 64 f16 = 4,096,000 B

    preproc_kernel<<<WAY + NQ, 256, 0, stream>>>(spt, qry, conv_w, bn_g, bn_b,
                                                 bn_m, bn_v, featT);
    corr_attn_kernel<<<NQ * WAY, 256, 0, stream>>>(spt, qry, featT, scale,
                                                   (float*)d_out);
}
